// PerformerAttention_54425825575070
// MI455X (gfx1250) — matrix operand order repaired and checked by reading, not compiled, not run on hardware
//
#include <hip/hip_runtime.h>
#include <hip/hip_bf16.h>

#define DEV __device__ __forceinline__

typedef __attribute__((ext_vector_type(16))) __bf16 v16bf;
typedef __attribute__((ext_vector_type(8)))  float  v8f;

union FragBf { unsigned int u[8]; v16bf v; };

// Problem constants (fixed by the reference)
constexpr int Bb   = 4;
constexpr int Nseq = 4096;
constexpr int Dm   = 1024;
constexpr int Hh   = 16;
constexpr int HD   = 64;
constexpr int Rr   = 256;
constexpr float EPSF       = 1e-6f;
constexpr float INV_SQRT_R = 0.0625f;   // 1/sqrt(256)

// ---- CDNA5 async global->LDS copy (ASYNCcnt path), with safe fallback ----
#if defined(__gfx1250__) && __has_builtin(__builtin_amdgcn_global_load_async_to_lds_b128)
#define GEMM_ASYNC 1
#else
#define GEMM_ASYNC 0
#endif

typedef int vec_i4 __attribute__((vector_size(16)));
typedef __attribute__((address_space(1))) vec_i4* gas_i4p;
typedef __attribute__((address_space(3))) vec_i4* las_i4p;

DEV void copy_b128_to_lds(const void* g, void* l) {
#if GEMM_ASYNC
  __builtin_amdgcn_global_load_async_to_lds_b128(
      (gas_i4p)(void*)g, (las_i4p)l, 0, 0);
#else
  *(uint4*)l = *(const uint4*)g;
#endif
}

DEV void wait_async_copies() {
#if GEMM_ASYNC
#if __has_builtin(__builtin_amdgcn_s_wait_asynccnt)
  __builtin_amdgcn_s_wait_asynccnt(0);
#else
  asm volatile("s_wait_asynccnt 0x0" ::: "memory");
#endif
#endif
}

DEV unsigned short f2bf(float f) {
  unsigned int u = __float_as_uint(f);
  u += 0x7FFFu + ((u >> 16) & 1u);      // round-to-nearest-even
  return (unsigned short)(u >> 16);
}
DEV float bf2f(unsigned int h16) { return __uint_as_float(h16 << 16); }

DEV float wave_sum(float s) {
#pragma unroll
  for (int m = 16; m; m >>= 1) s += __shfl_xor(s, m, 32);
  return s;
}

// ---------------------------------------------------------------- converts
__global__ void cvt_f32_bf16_kernel(const float* __restrict__ s,
                                    unsigned short* __restrict__ d, int n) {
  int i = blockIdx.x * blockDim.x + threadIdx.x;
  if (i < n) d[i] = f2bf(s[i]);
}

// projection (R,HD) f32 -> projT (HD,R) bf16
__global__ void cvt_projT_kernel(const float* __restrict__ s,
                                 unsigned short* __restrict__ d) {
  int i = blockIdx.x * blockDim.x + threadIdx.x;
  if (i < Rr * HD) {
    int r = i / HD, k = i % HD;
    d[k * Rr + r] = f2bf(s[i]);
  }
}

// ---------------------------------------------------------------- reductions
// 0.5*||row||^2 over rows of 64 bf16 (one wave per row, 1 uint/lane)
__global__ void halfnorm_kernel(const unsigned short* __restrict__ src,
                                float* __restrict__ out, int rows) {
  int gt = blockIdx.x * blockDim.x + threadIdx.x;
  int w = gt >> 5, lane = gt & 31;
  if (w >= rows) return;
  unsigned int u = ((const unsigned int*)(src + (long long)w * HD))[lane];
  float a = bf2f(u & 0xFFFFu), c = bf2f(u >> 16);
  float s = wave_sum(a * a + c * c);
  if (lane == 0) out[w] = 0.5f * s;
}

// k_sum[bh,r] = sum_n k_featT[bh,r,n]
__global__ void ksum_kernel(const unsigned short* __restrict__ kT,
                            float* __restrict__ ks, int rows) {
  int gt = blockIdx.x * blockDim.x + threadIdx.x;
  int w = gt >> 5, lane = gt & 31;
  if (w >= rows) return;
  const unsigned int* row = (const unsigned int*)(kT + (long long)w * Nseq);
  float s = 0.f;
  for (int i = lane; i < Nseq / 2; i += 32) {
    unsigned int u = row[i];
    s += bf2f(u & 0xFFFFu) + bf2f(u >> 16);
  }
  s = wave_sum(s);
  if (lane == 0) ks[w] = s;
}

// denom[bh,n] = dot(q_feat[bh,n,:], k_sum[bh,:])
__global__ void denom_kernel(const unsigned short* __restrict__ qf,
                             const float* __restrict__ ks,
                             float* __restrict__ dn, int rows) {
  int gt = blockIdx.x * blockDim.x + threadIdx.x;
  int w = gt >> 5, lane = gt & 31;
  if (w >= rows) return;
  const unsigned int* row = (const unsigned int*)(qf + (long long)w * Rr);
  const float* k = ks + (long long)(w >> 12) * Rr;   // bh = w / Nseq
  float s = 0.f;
  for (int i = lane; i < Rr / 2; i += 32) {
    unsigned int u = row[i];
    s += bf2f(u & 0xFFFFu) * k[2 * i] + bf2f(u >> 16) * k[2 * i + 1];
  }
  s = wave_sum(s);
  if (lane == 0) dn[w] = s;
}

// ---------------------------------------------------------------- GEMM
struct GP {
  const unsigned short* A;  // MxK bf16 row-major, lda
  const unsigned short* B;  // KxN bf16 row-major, ldb
  long long sAb, sBb;       // per-batch (blockIdx.z) element strides
  int lda, ldb, K;
  float* Cf;                // EPI 0/6
  unsigned short* Cb;       // EPI 2/3/4/5
  long long sCb; int ldc;
  const float* aux0;        // bias / halfnorm / denom
  unsigned short *dq, *dk, *dv;  // EPI 1
};

// EPI: 0=f32 row-major, 1=qkv scatter, 2=favor->q_feat, 3=favor->k_featT,
//      4=normalize+head scatter, 5=bf16 row-major, 6=f32+bias
template<int BM, int BN, int WR, int WC, int EPI>
__launch_bounds__(256)
__global__ void gemm_bf16_kernel(GP p) {
  constexpr int WM = BM / WR, WN = BN / WC;
  constexpr int MT = WM / 16, NT = WN / 16;
  static_assert(WR * WC == 8, "8 waves");
  static_assert(WM % 16 == 0 && WN % 16 == 0, "tiles");

  // double-buffered tiles
  __shared__ unsigned int ldsA[2][BM * 16];   // BM x 32 bf16 (K-pairs packed)
  __shared__ unsigned int ldsB[2][16 * BN];   // 16 K-pairs x BN cols

  const int bz = blockIdx.z;
  const unsigned short* A  = p.A + (long long)bz * p.sAb;
  const unsigned short* Bm = p.B + (long long)bz * p.sBb;
  const int m0 = blockIdx.y * BM, n0 = blockIdx.x * BN;
  const int tid = threadIdx.x, lane = tid & 31, wid = tid >> 5;
  const int wr = wid / WC, wc = wid % WC;
  const int lg = lane >> 4, ml = lane & 15;

  auto stage = [&](int k0, int buf) {
    // A tile: async global->LDS, 16B per op; row-major K pairs are natural
    for (int q = tid; q < BM * 4; q += 256) {
      int row = q >> 2, c4 = q & 3;
      const unsigned short* g = A + (long long)(m0 + row) * p.lda + k0 + (c4 << 3);
      copy_b128_to_lds(g, &ldsA[buf][(row << 4) + (c4 << 2)]);
    }
    // B tile: pack rows 2p,2p+1 into K-pair uints (WMMA B layout)
    for (int q = tid; q < 8 * BN; q += 256) {
      int pr = q / (BN / 2), j = q % (BN / 2);
      const unsigned int* b0 =
          (const unsigned int*)(Bm + (long long)(k0 + 2 * pr) * p.ldb + n0);
      const unsigned int* b1 =
          (const unsigned int*)(Bm + (long long)(k0 + 2 * pr + 1) * p.ldb + n0);
      unsigned int u0 = b0[j], u1 = b1[j];
      ldsB[buf][pr * BN + 2 * j]     = (u0 & 0xFFFFu) | (u1 << 16);
      ldsB[buf][pr * BN + 2 * j + 1] = (u0 >> 16) | (u1 & 0xFFFF0000u);
    }
  };

  v8f acc[MT][NT];
#pragma unroll
  for (int i = 0; i < MT; i++)
#pragma unroll
    for (int j = 0; j < NT; j++)
#pragma unroll
      for (int r = 0; r < 8; r++) acc[i][j][r] = 0.0f;

  stage(0, 0);
  wait_async_copies();
  __syncthreads();

  int buf = 0;
  for (int k0 = 0; k0 < p.K; k0 += 32) {
    const int nk = k0 + 32;
    if (nk < p.K) stage(nk, buf ^ 1);   // overlap next tile with compute

    FragBf a[MT], b[NT];
#pragma unroll
    for (int i = 0; i < MT; i++) {
      int Mr = wr * WM + i * 16 + ml;
#pragma unroll
      for (int v = 0; v < 8; v++) {
        int pr = (v & 3) + (lg << 2) + ((v >> 2) << 3);  // ISA 16-bit A layout
        a[i].u[v] = ldsA[buf][(Mr << 4) + pr];
      }
    }
#pragma unroll
    for (int j = 0; j < NT; j++) {
      int Nc = wc * WN + j * 16 + ml;
#pragma unroll
      for (int v = 0; v < 8; v++) {
        int pr = (v & 3) + (lg << 2) + ((v >> 2) << 3);  // same K order as A
        b[j].u[v] = ldsB[buf][pr * BN + Nc];
      }
    }
#pragma unroll
    for (int i = 0; i < MT; i++)
#pragma unroll
      for (int j = 0; j < NT; j++)
        acc[i][j] = __builtin_amdgcn_wmma_f32_16x16x32_bf16(
            false, a[i].v, false, b[j].v, (short)0, acc[i][j], false, false);

    if (nk < p.K) wait_async_copies();
    __syncthreads();
    buf ^= 1;
  }

#pragma unroll
  for (int i = 0; i < MT; i++) {
#pragma unroll
    for (int j = 0; j < NT; j++) {
#pragma unroll
      for (int r = 0; r < 8; r++) {
        int gm = m0 + wr * WM + i * 16 + r + (lg << 3);
        int gn = n0 + wc * WN + j * 16 + ml;
        float val = acc[i][j][r];
        if constexpr (EPI == 1) {          // qkv scatter -> (B,H,N,HD) bf16
          int bI = gm >> 12, n = gm & (Nseq - 1);
          int which = gn >> 10, h = (gn >> 6) & (Hh - 1), d = gn & (HD - 1);
          unsigned short* dst = which == 0 ? p.dq : (which == 1 ? p.dk : p.dv);
          dst[(((long long)(bI * Hh + h)) * Nseq + n) * HD + d] = f2bf(val);
        } else if constexpr (EPI == 2) {   // q favor map
          float f = __expf(val - p.aux0[(long long)bz * Nseq + gm]) * INV_SQRT_R + EPSF;
          p.Cb[((long long)bz * Nseq + gm) * Rr + gn] = f2bf(f);
        } else if constexpr (EPI == 3) {   // k favor map, transposed store
          float f = __expf(val - p.aux0[(long long)bz * Nseq + gm]) * INV_SQRT_R + EPSF;
          p.Cb[((long long)bz * Rr + gn) * Nseq + gm] = f2bf(f);
        } else if constexpr (EPI == 4) {   // normalize + head scatter
          float dnm = p.aux0[(long long)bz * Nseq + gm] + EPSF;
          int bI = bz >> 4, h = bz & (Hh - 1);
          p.Cb[((long long)(bI * Nseq + gm)) * Dm + h * HD + gn] = f2bf(val / dnm);
        } else if constexpr (EPI == 5) {   // bf16 row-major
          p.Cb[(long long)bz * p.sCb + (long long)gm * p.ldc + gn] = f2bf(val);
        } else if constexpr (EPI == 6) {   // f32 + bias
          p.Cf[(long long)gm * p.ldc + gn] = val + p.aux0[gn];
        } else {
          p.Cf[(long long)bz * p.sCb + (long long)gm * p.ldc + gn] = val;
        }
      }
    }
  }
}

// ---------------------------------------------------------------- launch
extern "C" void kernel_launch(void* const* d_in, const int* in_sizes, int n_in,
                              void* d_out, int out_size, void* d_ws, size_t ws_size,
                              hipStream_t stream) {
  const float* x      = (const float*)d_in[0];
  const float* w_qkv  = (const float*)d_in[1];
  const float* w_proj = (const float*)d_in[2];
  const float* b_proj = (const float*)d_in[3];
  const float* proj   = (const float*)d_in[4];
  float* out = (float*)d_out;

  char* ws = (char*)d_ws;
  size_t off = 0;
  auto alloc = [&](size_t bytes) -> char* {
    size_t o = (off + 255) & ~(size_t)255;
    off = o + bytes;
    return ws + o;
  };
  const long long BHN = (long long)Bb * Hh * Nseq;
  unsigned short* x_bf     = (unsigned short*)alloc((size_t)Bb * Nseq * Dm * 2);
  unsigned short* wqkv_bf  = (unsigned short*)alloc((size_t)Dm * 3 * Dm * 2);
  unsigned short* wproj_bf = (unsigned short*)alloc((size_t)Dm * Dm * 2);
  unsigned short* projT_bf = (unsigned short*)alloc((size_t)HD * Rr * 2);
  unsigned short* qb       = (unsigned short*)alloc((size_t)BHN * HD * 2);
  unsigned short* kb       = (unsigned short*)alloc((size_t)BHN * HD * 2);
  unsigned short* vb       = (unsigned short*)alloc((size_t)BHN * HD * 2);
  float*          nq       = (float*)alloc((size_t)BHN * 4);
  float*          nk       = (float*)alloc((size_t)BHN * 4);
  unsigned short* qfeat    = (unsigned short*)alloc((size_t)BHN * Rr * 2);
  unsigned short* kfeatT   = (unsigned short*)alloc((size_t)BHN * Rr * 2);
  unsigned short* kvb      = (unsigned short*)alloc((size_t)Bb * Hh * Rr * HD * 2);
  float*          ksum     = (float*)alloc((size_t)Bb * Hh * Rr * 4);
  float*          dnm      = (float*)alloc((size_t)BHN * 4);
  unsigned short* attn     = (unsigned short*)alloc((size_t)Bb * Nseq * Dm * 2);
  if (off > ws_size) return;  // workspace too small: bail deterministically

  // 1) convert inputs to bf16
  {
    int n = Bb * Nseq * Dm;
    cvt_f32_bf16_kernel<<<(n + 255) / 256, 256, 0, stream>>>(x, x_bf, n);
    n = Dm * 3 * Dm;
    cvt_f32_bf16_kernel<<<(n + 255) / 256, 256, 0, stream>>>(w_qkv, wqkv_bf, n);
    n = Dm * Dm;
    cvt_f32_bf16_kernel<<<(n + 255) / 256, 256, 0, stream>>>(w_proj, wproj_bf, n);
    cvt_projT_kernel<<<(Rr * HD + 255) / 256, 256, 0, stream>>>(proj, projT_bf);
  }

  // 2) qkv = x @ w_qkv, scattered to per-head q/k/v (M=16384,N=3072,K=1024)
  {
    GP g{}; g.A = x_bf; g.B = wqkv_bf; g.lda = Dm; g.ldb = 3 * Dm; g.K = Dm;
    g.dq = qb; g.dk = kb; g.dv = vb;
    gemm_bf16_kernel<128, 128, 2, 4, 1>
        <<<dim3(3 * Dm / 128, (Bb * Nseq) / 128, 1), 256, 0, stream>>>(g);
  }

  // 3) half squared norms of q, k rows
  {
    int rows = (int)BHN;
    halfnorm_kernel<<<(rows * 32 + 255) / 256, 256, 0, stream>>>(qb, nq, rows);
    halfnorm_kernel<<<(rows * 32 + 255) / 256, 256, 0, stream>>>(kb, nk, rows);
  }

  // 4) favor maps: per-(b,h) GEMM (M=4096, N=256, K=64) + exp epilogue
  {
    GP g{}; g.B = projT_bf; g.sBb = 0; g.ldb = Rr; g.K = HD; g.lda = HD;
    g.sAb = (long long)Nseq * HD;
    g.A = qb; g.aux0 = nq; g.Cb = qfeat;
    gemm_bf16_kernel<128, 128, 2, 4, 2>
        <<<dim3(Rr / 128, Nseq / 128, Bb * Hh), 256, 0, stream>>>(g);
    g.A = kb; g.aux0 = nk; g.Cb = kfeatT;
    gemm_bf16_kernel<128, 128, 2, 4, 3>
        <<<dim3(Rr / 128, Nseq / 128, Bb * Hh), 256, 0, stream>>>(g);
  }

  // 5) k_sum[bh,r]
  {
    int rows = Bb * Hh * Rr;
    ksum_kernel<<<(rows * 32 + 255) / 256, 256, 0, stream>>>(kfeatT, ksum, rows);
  }

  // 6) kv[bh] = k_featT @ v  (M=256, N=64, K=4096) -> bf16
  {
    GP g{}; g.A = kfeatT; g.sAb = (long long)Rr * Nseq; g.lda = Nseq;
    g.B = vb; g.sBb = (long long)Nseq * HD; g.ldb = HD; g.K = Nseq;
    g.Cb = kvb; g.ldc = HD; g.sCb = (long long)Rr * HD;
    gemm_bf16_kernel<128, 64, 4, 2, 5>
        <<<dim3(1, Rr / 128, Bb * Hh), 256, 0, stream>>>(g);
  }

  // 7) denom[bh,n] = q_feat . k_sum
  {
    int rows = (int)BHN;
    denom_kernel<<<(rows * 32 + 255) / 256, 256, 0, stream>>>(qfeat, ksum, dnm, rows);
  }

  // 8) out[bh] = (q_feat @ kv) / (denom+eps), scattered to (B,N,D) bf16
  {
    GP g{}; g.A = qfeat; g.sAb = (long long)Nseq * Rr; g.lda = Rr;
    g.B = kvb; g.sBb = (long long)Rr * HD; g.ldb = HD; g.K = Rr;
    g.aux0 = dnm; g.Cb = attn;
    gemm_bf16_kernel<128, 64, 4, 2, 4>
        <<<dim3(1, Nseq / 128, Bb * Hh), 256, 0, stream>>>(g);
  }

  // 9) final: attn @ w_proj + b_proj -> f32 d_out (M=16384, N=1024, K=1024)
  {
    GP g{}; g.A = attn; g.lda = Dm; g.B = wproj_bf; g.ldb = Dm; g.K = Dm;
    g.Cf = out; g.ldc = Dm; g.aux0 = b_proj;
    gemm_bf16_kernel<128, 128, 2, 4, 6>
        <<<dim3(Dm / 128, (Bb * Nseq) / 128, 1), 256, 0, stream>>>(g);
  }
}
